// StochasticTernaryLinear_38998303048322
// MI455X (gfx1250) — matrix operand order repaired and checked by reading, not compiled, not run on hardware
//
#include <hip/hip_runtime.h>

typedef __attribute__((ext_vector_type(16))) _Float16     v16h;
typedef __attribute__((ext_vector_type(8)))  float        v8f;
typedef __attribute__((ext_vector_type(4)))  float        v4f;
typedef __attribute__((ext_vector_type(4)))  unsigned int v4u;

union HalfPack { unsigned int u; _Float16 h[2]; };
union BFrag    { v4u u4[2];      v16h h;        };

// LDS B-fragment store: 8 ntile * 4 ktile * 32 lane * 8 j  (each slot = 2 packed f16)
#define LDSB_WORDS 8192

__global__ __launch_bounds__(256)
void ternary_linear_wmma(const float* __restrict__ x,
                         const float* __restrict__ wp,
                         const float* __restrict__ wn,
                         const float* __restrict__ bias,
                         float* __restrict__ out) {
  __shared__ unsigned int ldsB[LDSB_WORDS];

  const int tid  = threadIdx.x;
  const int lane = tid & 31;
  const int wave = tid >> 5;

  // ---------------- Phase 1: ternarize weights into WMMA B-fragment layout ----------------
  // B (32x16, f16) per ISA 7.12.2: VGPR j<4, lane<16 -> N=lane, K = 2j, 2j+1; j>=4 -> K = 16+2(j-4), +1
  //                                         lane>=16 -> N=lane-16, K offset +8
  // LDS address: ((ntile*4 + ktile)*32 + lane)*8 + j   (dwords), so per-lane j=0..7 is
  // 32 contiguous bytes -> two ds_load_b128 per fragment.
#pragma unroll
  for (int slot = tid; slot < LDSB_WORDS; slot += 256) {
    const int j     = slot & 7;
    const int l     = (slot >> 3) & 31;
    const int ktile = (slot >> 8) & 3;
    const int ntile = (slot >> 10) & 7;
    const int k0 = ktile * 32 + ((j & 4) << 2) + ((l & 16) >> 1) + 2 * (j & 3);      // table order: lanes 16-31 at K+8
    const int n  = ntile * 16 + (l & 15);
    // B[k][n] = w.T[k][n] = w[n][k];  w = (sign(p) - sign(n)) * 0.5  in {-1, 0, +1}
    const float p0 = wp[n * 128 + k0];
    const float p1 = wp[n * 128 + k0 + 1];
    const float q0 = wn[n * 128 + k0];
    const float q1 = wn[n * 128 + k0 + 1];
    const float w0 = 0.5f * ((p0 >= 0.0f ? 1.0f : -1.0f) - (q0 >= 0.0f ? 1.0f : -1.0f));
    const float w1 = 0.5f * ((p1 >= 0.0f ? 1.0f : -1.0f) - (q1 >= 0.0f ? 1.0f : -1.0f));
    HalfPack hp;
    hp.h[0] = (_Float16)w0;   // exact: {-1,0,+1}
    hp.h[1] = (_Float16)w1;
    ldsB[slot] = hp.u;
  }
  __syncthreads();

  // ---------------- Phase 2: WMMA GEMM, 32 rows per wave (two 16-row M tiles) ----------------
  const int rowBlk = blockIdx.x * 256 + wave * 32;     // this wave's 32 rows
  const int mrow   = lane & 15;                        // row within a 16-row tile
  const int kb_off = (lane & 16) >> 1;                 // A layout: lanes 0-15 start K+0, 16-31 K+8

  // Hoist all bias values (C init) so no global load sits in the hot loop.
  float bvals[8];
#pragma unroll
  for (int nt = 0; nt < 8; ++nt) bvals[nt] = bias[nt * 16 + mrow];

  // Load + convert all A fragments (x is streamed once -> non-temporal loads).
  v16h a0[4], a1[4];
  {
    const float* xr0 = x + (size_t)(rowBlk + mrow) * 128;
    const float* xr1 = xr0 + (size_t)16 * 128;
#pragma unroll
    for (int kt = 0; kt < 4; ++kt) {
      const int kb = kt * 32 + kb_off;
      const v4f f0 = __builtin_nontemporal_load((const v4f*)(xr0 + kb));
      const v4f f1 = __builtin_nontemporal_load((const v4f*)(xr0 + kb + 4));
      const v4f f2 = __builtin_nontemporal_load((const v4f*)(xr0 + kb + 16));
      const v4f f3 = __builtin_nontemporal_load((const v4f*)(xr0 + kb + 20));
      v16h a;
      a[0]  = (_Float16)f0[0]; a[1]  = (_Float16)f0[1]; a[2]  = (_Float16)f0[2]; a[3]  = (_Float16)f0[3];
      a[4]  = (_Float16)f1[0]; a[5]  = (_Float16)f1[1]; a[6]  = (_Float16)f1[2]; a[7]  = (_Float16)f1[3];
      a[8]  = (_Float16)f2[0]; a[9]  = (_Float16)f2[1]; a[10] = (_Float16)f2[2]; a[11] = (_Float16)f2[3];
      a[12] = (_Float16)f3[0]; a[13] = (_Float16)f3[1]; a[14] = (_Float16)f3[2]; a[15] = (_Float16)f3[3];
      a0[kt] = a;

      const v4f g0 = __builtin_nontemporal_load((const v4f*)(xr1 + kb));
      const v4f g1 = __builtin_nontemporal_load((const v4f*)(xr1 + kb + 4));
      const v4f g2 = __builtin_nontemporal_load((const v4f*)(xr1 + kb + 16));
      const v4f g3 = __builtin_nontemporal_load((const v4f*)(xr1 + kb + 20));
      v16h b;
      b[0]  = (_Float16)g0[0]; b[1]  = (_Float16)g0[1]; b[2]  = (_Float16)g0[2]; b[3]  = (_Float16)g0[3];
      b[4]  = (_Float16)g1[0]; b[5]  = (_Float16)g1[1]; b[6]  = (_Float16)g1[2]; b[7]  = (_Float16)g1[3];
      b[8]  = (_Float16)g2[0]; b[9]  = (_Float16)g2[1]; b[10] = (_Float16)g2[2]; b[11] = (_Float16)g2[3];
      b[12] = (_Float16)g3[0]; b[13] = (_Float16)g3[1]; b[14] = (_Float16)g3[2]; b[15] = (_Float16)g3[3];
      a1[kt] = b;
    }
  }

  const int vrow_base = (lane & 16) ? 8 : 0;           // C/D layout: lanes 16-31 hold M=8..15

#pragma unroll
  for (int ntile = 0; ntile < 8; ++ntile) {
    const unsigned int* bbase = &ldsB[(unsigned)((ntile * 4) * 32 + lane) * 8];

    // Preload all 4 K-step B fragments so the compiler can clause 8 ds_load_b128
    // and overlap LDS latency with WMMA issue instead of wait-per-pair.
    BFrag bf[4];
#pragma unroll
    for (int kt = 0; kt < 4; ++kt) {
      bf[kt].u4[0] = *(const v4u*)(bbase + kt * 256);
      bf[kt].u4[1] = *(const v4u*)(bbase + kt * 256 + 4);
    }

    // C init = bias broadcast: each acc VGPR has fixed N = ntile*16 + (lane&15)
    const float bv = bvals[ntile];
    v8f c0 = { bv, bv, bv, bv, bv, bv, bv, bv };
    v8f c1 = c0;

#pragma unroll
    for (int kt = 0; kt < 4; ++kt) {
      c0 = __builtin_amdgcn_wmma_f32_16x16x32_f16(false, a0[kt], false, bf[kt].h,
                                                  (short)0, c0, false, false);
      c1 = __builtin_amdgcn_wmma_f32_16x16x32_f16(false, a1[kt], false, bf[kt].h,
                                                  (short)0, c1, false, false);
    }

    // Store: VGPR v -> row (vrow_base + v), col = ntile*16 + (lane&15). Streamed -> NT stores.
    float* o = out + (size_t)rowBlk * 128 + (size_t)(ntile * 16 + mrow);
#pragma unroll
    for (int v = 0; v < 8; ++v) {
      __builtin_nontemporal_store(c0[v], o + (size_t)(vrow_base + v) * 128);
      __builtin_nontemporal_store(c1[v], o + (size_t)(16 + vrow_base + v) * 128);
    }
  }
}

extern "C" void kernel_launch(void* const* d_in, const int* in_sizes, int n_in,
                              void* d_out, int out_size, void* d_ws, size_t ws_size,
                              hipStream_t stream) {
  (void)n_in; (void)out_size; (void)d_ws; (void)ws_size;
  const float* x    = (const float*)d_in[0];
  const float* wp   = (const float*)d_in[1];
  const float* wn   = (const float*)d_in[2];
  const float* bias = (const float*)d_in[3];
  float* out = (float*)d_out;

  const int rows = in_sizes[0] / 128;   // B = 524288
  const int grid = rows / 256;          // 256 rows per block (8 waves x 32 rows)
  ternary_linear_wmma<<<grid, 256, 0, stream>>>(x, wp, wn, bias, out);
}
